// Temporal_EncDec_V1_11501922419165
// MI455X (gfx1250) — matrix operand order repaired and checked by reading, not compiled, not run on hardware
//
#include <hip/hip_runtime.h>
#include <cstdint>

typedef __attribute__((ext_vector_type(16))) _Float16 v16h;
typedef __attribute__((ext_vector_type(8)))  float    v8f;

#define HH 96
#define WW 96
#define HWSZ (HH*WW)
#define PADW 98
#define MAXCP 40   // max padded channel stride

// ---------------------------------------------------------------------------
// Pre-shuffle f32 conv weights (OIHW) into the exact WMMA 16x32 f16 A-fragment
// VGPR layout, with K ordered as k = (ky*3+kx)*Cin + c and zero-padded to a
// multiple of 32. frag index == ((ks*nM + mt)*32 + lane)*8 + v (packed f16x2).
// ---------------------------------------------------------------------------
__global__ __launch_bounds__(256) void prep_weights(const float* __restrict__ Wsrc,
                                                    unsigned int* __restrict__ frag,
                                                    int Cin, int nM, int total)
{
    int tid = blockIdx.x * blockDim.x + threadIdx.x;
    if (tid >= total) return;
    int v    = tid & 7;
    int lane = (tid >> 3) & 31;
    int mt   = (tid >> 8) % nM;
    int ks   = tid / (nM << 8);
    int m  = mt * 16 + (lane & 15);
    // 16-bit A 16x32 layout (contiguous): lanes 0-15: V0-7 K=0..15;
    //                                     lanes 16-31: V0-7 K=16..31
    int kk = ((lane < 16) ? 0 : 16) + 2 * v;
    int kg = ks * 32 + kk;
    unsigned int packed = 0u;
    int Ktot = Cin * 9;
    if (kg < Ktot) {
        int j  = kg / Cin;
        int c  = kg - j * Cin;          // even (kg even, Cin even)
        int ky = j / 3, kx = j - ky * 3;
        float w0 = Wsrc[((m * Cin + c    ) * 3 + ky) * 3 + kx];
        float w1 = Wsrc[((m * Cin + c + 1) * 3 + ky) * 3 + kx];
        _Float16 h0 = (_Float16)w0, h1 = (_Float16)w1;
        unsigned short u0 = *(unsigned short*)&h0;
        unsigned short u1 = *(unsigned short*)&h1;
        packed = (unsigned int)u0 | ((unsigned int)u1 << 16);
    }
    frag[tid] = packed;
}

// compile-time B-gather DWORD offset for global k index kg
// (returns 0 for K-padding region: A fragment is zero there, so any in-range
//  finite LDS value multiplies to zero)
__host__ __device__ constexpr int bOffDw(int kg, int Cin, int CinP)
{
    return (kg >= Cin * 9)
        ? 0
        : (((kg / Cin / 3) * PADW + (kg / Cin % 3)) * CinP + (kg % Cin)) / 2;
}

// ---------------------------------------------------------------------------
// One ConvLSTM timestep for one layer (fully specialized).
// Block = (row y, batch b), 256 threads = 8 waves (wave32).
// Gate conv as WMMA GEMM: M = 4*Ch (nM 16-tiles), N = 96 (6 tiles), K = Cin*9.
// Each wave owns one M-tile and NTPW N-tiles: one shared A fragment feeds
// NTPW independent accumulator chains (fills WMMA hazard slots, cuts A loads).
// For Cin=32 the B fragment is two aligned ds_load_b128 (CinP=40 keeps runs
// 16B-aligned and bank-conflict-free); Cin=24 uses paired b32 gathers.
// ---------------------------------------------------------------------------
template <int Cx, int Ch, int Cin, int Ksteps, int nM>
__global__ __launch_bounds__(256) void convlstm_step(
    const float* __restrict__ xBase, long long xBS,
    const float* __restrict__ hPrev, long long hBS,
    float* __restrict__ cState,
    float* __restrict__ outH,
    const float* __restrict__ bias,
    const unsigned int* __restrict__ frag,
    int firstT,
    float* __restrict__ hFinal, float* __restrict__ cFinal)
{
    constexpr int CinP = (Cin == 32) ? 40 : (Cin + 2); // padded channel stride
    __shared__ _Float16 sIn[3 * PADW * MAXCP];  // [ky][x'][c], stride CinP
    __shared__ float    sZ[64 * WW];            // gate pre-activations

    const int y   = blockIdx.x;
    const int b   = blockIdx.y;
    const int tid = threadIdx.x;

    // ---- stage rows y-1..y+1 of cat([x, h]) as f16, channel-interleaved ----
    constexpr int nStage = 3 * PADW * Cin;
    for (int i = tid; i < nStage; i += 256) {
        int c  = i % Cin;
        int r  = i / Cin;
        int xp = r % PADW;
        int ky = r / PADW;
        int yy = y + ky - 1;
        int xx = xp - 1;
        float v = 0.f;
        if (yy >= 0 && yy < HH && xx >= 0 && xx < WW) {
            if (c < Cx)
                v = xBase[(long long)b * xBS + (long long)c * HWSZ + yy * WW + xx];
            else if (hPrev)
                v = hPrev[(long long)b * hBS + (long long)(c - Cx) * HWSZ + yy * WW + xx];
        }
        sIn[(ky * PADW + xp) * CinP + c] = (_Float16)v;
    }
    __syncthreads();

    // ---- WMMA GEMM ----
    const int wave    = tid >> 5;
    const int lane    = tid & 31;
    const int laneN   = lane & 15;
    const bool hiHalf = (lane >= 16);

    constexpr int NTPW = (nM == 4) ? 3 : 2;       // N-tiles per wave
    constexpr int WUSE = nM * (6 / NTPW);         // active waves (8 or 6)

    if (wave < WUSE) {
        const int mt  = wave % nM;
        const int nt0 = (wave / nM) * NTPW;

        int xcol[NTPW];
        const unsigned int* bp[NTPW];
        #pragma unroll
        for (int j = 0; j < NTPW; ++j) {
            xcol[j] = (nt0 + j) * 16 + laneN;
            bp[j] = (const unsigned int*)((const char*)sIn +
                                          (size_t)xcol[j] * (CinP * 2));
        }
        const int hsel = hiHalf ? 8 : 0;  // hi half: +8 dwords (c 16..31)

        // per-M-tile A-fragment base; k-step strides are immediate offsets
        const unsigned int* fpBase = frag + ((size_t)mt * 32 + lane) * 8;
        __builtin_prefetch((const void*)fpBase, 0, 1);

        v8f acc[NTPW];
        #pragma unroll
        for (int j = 0; j < NTPW; ++j)
            acc[j] = (v8f){0.f,0.f,0.f,0.f,0.f,0.f,0.f,0.f};

        #pragma unroll
        for (int ks = 0; ks < Ksteps; ++ks) {
            // shared A fragment: pre-shuffled f16 weights, two b128 loads
            union { uint4 q[2]; v16h h; } A;
            A.q[0] = *(const uint4*)(fpBase + ks * nM * 256);
            A.q[1] = *(const uint4*)(fpBase + ks * nM * 256 + 4);

            // NTPW B fragments
            union BFr { unsigned int u[8]; uint4 q[2]; v16h h; };
            BFr Bv[NTPW];
            if constexpr (Cin == 32) {
                // 16-k half-run = 8 contiguous, 16B-aligned LDS dwords
                #pragma unroll
                for (int j = 0; j < NTPW; ++j) {
                    const int dwLo = bOffDw(32 * ks, Cin, CinP);
                    const uint4* p = (const uint4*)(bp[j] + dwLo + hsel);
                    Bv[j].q[0] = p[0];
                    Bv[j].q[1] = p[1];
                }
            } else {
                #pragma unroll
                for (int v = 0; v < 8; ++v) {
                    const int offLo = bOffDw((2 * v)      + ks * 32, Cin, CinP);
                    const int offHi = bOffDw((2 * v + 16) + ks * 32, Cin, CinP);
                    const int sel   = hiHalf ? offHi : offLo;
                    #pragma unroll
                    for (int j = 0; j < NTPW; ++j)
                        Bv[j].u[v] = bp[j][sel];
                }
            }

            // independent accumulator chains (no D->A/B hazard between them)
            #pragma unroll
            for (int j = 0; j < NTPW; ++j)
                acc[j] = __builtin_amdgcn_wmma_f32_16x16x32_f16(
                             false, A.h, false, Bv[j].h,
                             (short)0, acc[j], false, false);
        }

        // C/D layout: VGPR r -> lanes0-15: M=r, lanes16-31: M=8+r; N = lane&15
        const int mrow = mt * 16 + (hiHalf ? 8 : 0);
        #pragma unroll
        for (int j = 0; j < NTPW; ++j)
            #pragma unroll
            for (int r = 0; r < 8; ++r)
                sZ[(mrow + r) * WW + xcol[j]] = acc[j][r];
    }
    __syncthreads();

    // ---- pointwise LSTM update ----
    constexpr int hid = Ch;
    const long long cBS = (long long)Ch * HWSZ;
    for (int i = tid; i < hid * WW; i += 256) {
        int ch = i / WW;
        int xx = i - ch * WW;
        float iv = sZ[ ch            * WW + xx] + bias[ch];
        float fv = sZ[(hid   + ch)   * WW + xx] + bias[hid + ch];
        float ov = sZ[(2*hid + ch)   * WW + xx] + bias[2 * hid + ch];
        float gv = sZ[(3*hid + ch)   * WW + xx] + bias[3 * hid + ch];
        long long soff = (long long)b * cBS + (long long)ch * HWSZ + y * WW + xx;
        float cp = firstT ? 0.f : cState[soff];
        float si = 1.f / (1.f + __expf(-iv));
        float sf = 1.f / (1.f + __expf(-fv));
        float so = 1.f / (1.f + __expf(-ov));
        float cn = sf * cp + si * tanhf(gv);
        float hn = so * tanhf(cn);
        cState[soff] = cn;
        outH[(long long)b * hBS + (long long)ch * HWSZ + y * WW + xx] = hn;
        if (hFinal) { hFinal[soff] = hn; cFinal[soff] = cn; }
    }
}

// ---------------------------------------------------------------------------
extern "C" void kernel_launch(void* const* d_in, const int* in_sizes, int n_in,
                              void* d_out, int out_size, void* d_ws, size_t ws_size,
                              hipStream_t stream)
{
    const float* x  = (const float*)d_in[0];
    const float* W1 = (const float*)d_in[1];
    const float* b1 = (const float*)d_in[2];
    const float* W2 = (const float*)d_in[3];
    const float* b2 = (const float*)d_in[4];
    const float* W3 = (const float*)d_in[5];
    const float* b3 = (const float*)d_in[6];

    const long long HW = HWSZ;
    const int T = 24, B = 8;

    float* out3 = (float*)d_out;                       // (8,24,16,96,96)
    float* h2o  = out3 + (long long)B * T * 16 * HW;   // (8,8,96,96)
    float* c2o  = h2o  + (long long)B * 8 * HW;        // (8,8,96,96)

    char* ws = (char*)d_ws;
    size_t off = 0;
    auto carve = [&](size_t bytes) -> void* {
        void* p = ws + off;
        off += (bytes + 255) & ~(size_t)255;
        return p;
    };
    float* out1      = (float*)carve((size_t)B * T * 16 * HW * 4);
    float* out2      = (float*)carve((size_t)B * T * 8  * HW * 4);
    float* c1        = (float*)carve((size_t)B * 16 * HW * 4);
    float* c2s       = (float*)carve((size_t)B * 8  * HW * 4);
    float* c3        = (float*)carve((size_t)B * 16 * HW * 4);
    unsigned int* f1 = (unsigned int*)carve((size_t)9 * 4 * 256 * 4);
    unsigned int* f2 = (unsigned int*)carve((size_t)7 * 2 * 256 * 4);
    unsigned int* f3 = (unsigned int*)carve((size_t)7 * 4 * 256 * 4);

    prep_weights<<<(9*4*256 + 255) / 256, 256, 0, stream>>>(W1, f1, 32, 4, 9*4*256);
    prep_weights<<<(7*2*256 + 255) / 256, 256, 0, stream>>>(W2, f2, 24, 2, 7*2*256);
    prep_weights<<<(7*4*256 + 255) / 256, 256, 0, stream>>>(W3, f3, 24, 4, 7*4*256);

    dim3 grid(HH, B), blk(256);
    for (int t = 0; t < T; ++t) {
        // Layer 1: Cx=16, hid=16, Cin=32, K=288 (9 k-steps), M=64 (4 tiles)
        convlstm_step<16, 16, 32, 9, 4><<<grid, blk, 0, stream>>>(
            x + (long long)t * 16 * HW, (long long)T * 16 * HW,
            t ? out1 + (long long)(t - 1) * 16 * HW : nullptr, (long long)T * 16 * HW,
            c1, out1 + (long long)t * 16 * HW, b1, f1,
            t == 0, nullptr, nullptr);
        // Layer 2: Cx=16, hid=8, Cin=24, K=216 pad 224 (7 k-steps), M=32 (2 tiles)
        convlstm_step<16, 8, 24, 7, 2><<<grid, blk, 0, stream>>>(
            out1 + (long long)t * 16 * HW, (long long)T * 16 * HW,
            t ? out2 + (long long)(t - 1) * 8 * HW : nullptr, (long long)T * 8 * HW,
            c2s, out2 + (long long)t * 8 * HW, b2, f2,
            t == 0,
            (t == T - 1) ? h2o : nullptr, (t == T - 1) ? c2o : nullptr);
        // Layer 3: Cx=8, hid=16, Cin=24, K=216 pad 224 (7 k-steps), M=64 (4 tiles)
        convlstm_step<8, 16, 24, 7, 4><<<grid, blk, 0, stream>>>(
            out2 + (long long)t * 8 * HW, (long long)T * 8 * HW,
            t ? out3 + (long long)(t - 1) * 16 * HW : nullptr, (long long)T * 16 * HW,
            c3, out3 + (long long)t * 16 * HW, b3, f3,
            t == 0, nullptr, nullptr);
    }
    (void)in_sizes; (void)n_in; (void)out_size; (void)ws_size;
}
